// Qwen3SimpleMoE_31636729102462
// MI455X (gfx1250) — compile-verified, matrix operand order checked by reading, not run on hardware
//
#include <hip/hip_runtime.h>
#include <hip/hip_bf16.h>

// Problem constants (from reference): B=2, S=2048, D=1024, E=8, F=2816, K=2
#define TOKENS 4096
#define DDIM   1024
#define NEXP   8
#define FDIM   2816

typedef __bf16 bf16;
typedef __attribute__((ext_vector_type(16))) __bf16 v16bf;
typedef __attribute__((ext_vector_type(8)))  __bf16 v8bf;
typedef __attribute__((ext_vector_type(8)))  float  v8f;
typedef __attribute__((ext_vector_type(4)))  int    v4i;

#define WMMA_BF16(a, b, c) \
  __builtin_amdgcn_wmma_f32_16x16x32_bf16(false, (a), false, (b), (short)0, (c), false, false)

// ---- CDNA5 async global->LDS staging (ASYNCcnt-tracked), with fallback ----
#if __has_builtin(__builtin_amdgcn_global_load_async_to_lds_b128)
#define USE_ASYNC_LDS 1
typedef __attribute__((address_space(1))) void as1_void;
typedef __attribute__((address_space(3))) void as3_void;
typedef __attribute__((address_space(1))) v4i  as1_v4i;
typedef __attribute__((address_space(3))) v4i  as3_v4i;
__device__ __forceinline__ void async_ld16(void* lds, const void* g) {
  __builtin_amdgcn_global_load_async_to_lds_b128(
      (as1_v4i*)(as1_void*)(void*)g,
      (as3_v4i*)(as3_void*)lds, 0, 0);
}
#if __has_builtin(__builtin_amdgcn_s_wait_asynccnt)
#define WAIT_ASYNC() __builtin_amdgcn_s_wait_asynccnt(0)
#else
#define WAIT_ASYNC() asm volatile("s_wait_asynccnt 0x0" ::: "memory")
#endif
#else
#define USE_ASYNC_LDS 0
#endif

// ---------------------------------------------------------------- utilities

__global__ void k_zero_f32(float* __restrict__ p, int n) {
  int i = blockIdx.x * blockDim.x + threadIdx.x;
  if (i < n) p[i] = 0.0f;
}

__global__ void k_cvt_bf16(const float* __restrict__ s, bf16* __restrict__ d, int n) {
  int i = (blockIdx.x * blockDim.x + threadIdx.x) * 4;
  if (i >= n) return;
  float4 v = *(const float4*)(s + i);
  union { bf16 b[4]; unsigned long long u; } o;
  o.b[0] = (bf16)v.x; o.b[1] = (bf16)v.y; o.b[2] = (bf16)v.z; o.b[3] = (bf16)v.w;
  *(unsigned long long*)(d + i) = o.u;
}

// fp32 [K,N] -> bf16 [N,K] tiled transpose; blockIdx.z = matrix index (stride K*N).
__global__ __launch_bounds__(256)
void k_cvt_t(const float* __restrict__ src, bf16* __restrict__ dst, int K, int N) {
  __shared__ bf16 tile[32][33];
  size_t moff = (size_t)blockIdx.z * K * N;
  int nb = blockIdx.x * 32, kb = blockIdx.y * 32;
  int tx = threadIdx.x, ty = threadIdx.y;       // (32, 8)
#pragma unroll
  for (int i = 0; i < 4; ++i) {
    int k = kb + ty + i * 8;
    tile[ty + i * 8][tx] = (bf16)src[moff + (size_t)k * N + nb + tx];
  }
  __syncthreads();
#pragma unroll
  for (int i = 0; i < 4; ++i) {
    int n = nb + ty + i * 8;
    dst[moff + (size_t)n * K + kb + tx] = tile[tx][ty + i * 8];
  }
}

// ---------------------------------------------------------------- router

__global__ void k_router(const float* __restrict__ x, const float* __restrict__ gw,
                         int* __restrict__ tki, float* __restrict__ tkw,
                         int* __restrict__ cnt) {
  int t = blockIdx.x * blockDim.x + threadIdx.x;
  if (t >= TOKENS) return;
  float acc[NEXP];
#pragma unroll
  for (int e = 0; e < NEXP; ++e) acc[e] = 0.0f;
  const float* xr = x + (size_t)t * DDIM;
  for (int d = 0; d < DDIM; ++d) {
    float xv = xr[d];
#pragma unroll
    for (int e = 0; e < NEXP; ++e) acc[e] = fmaf(xv, gw[d * NEXP + e], acc[e]);
  }
  int i0 = 0; float m0 = acc[0];
#pragma unroll
  for (int e = 1; e < NEXP; ++e) if (acc[e] > m0) { m0 = acc[e]; i0 = e; }
  int i1 = -1; float m1 = -3.4e38f;
#pragma unroll
  for (int e = 0; e < NEXP; ++e) if (e != i0 && acc[e] > m1) { m1 = acc[e]; i1 = e; }
  float p1 = 1.0f / (1.0f + __expf(m0 - m1));
  float p0 = 1.0f - p1;
  tki[t * 2 + 0] = i0; tki[t * 2 + 1] = i1;
  tkw[t * 2 + 0] = p0; tkw[t * 2 + 1] = p1;
  atomicAdd(&cnt[i0], 1);
  atomicAdd(&cnt[i1], 1);
}

__global__ void k_scan(const int* __restrict__ cnt, int* __restrict__ offs,
                       int* __restrict__ cursor) {
  if (threadIdx.x == 0 && blockIdx.x == 0) {
    int s = 0;
    for (int e = 0; e < NEXP; ++e) { offs[e] = s; cursor[e] = s; s += cnt[e]; }
  }
}

__global__ void k_scatter(const int* __restrict__ tki, const float* __restrict__ tkw,
                          int* __restrict__ cursor, int* __restrict__ list,
                          float* __restrict__ wlist) {
  int t = blockIdx.x * blockDim.x + threadIdx.x;
  if (t >= TOKENS) return;
#pragma unroll
  for (int k = 0; k < 2; ++k) {
    int e = tki[t * 2 + k];
    int pos = atomicAdd(&cursor[e], 1);
    list[pos] = t;
    wlist[pos] = tkw[t * 2 + k];
  }
}

// ---------------------------------------------------------------- frag helpers

__device__ __forceinline__ v16bf ldfrag(const bf16* p) {
  union { v16bf v; v8bf h[2]; } u;
  u.h[0] = *(const v8bf*)(p);
  u.h[1] = *(const v8bf*)(p + 16);
  return u.v;
}

__device__ __forceinline__ float silu_f(float g) {
  return g * __builtin_amdgcn_rcpf(1.0f + __expf(-g));
}

// ---------------------------------------------------------------- grouped GEMM 1
// act = silu(X @ Wg) * (X @ Wu); BM=128, BN=64, BK=32; 8 waves, 32x32 per wave.
__global__ __launch_bounds__(256)
void k_gemm_gateup(const bf16* __restrict__ xb,
                   const bf16* __restrict__ wgT, const bf16* __restrict__ wuT,
                   const bf16* __restrict__ wsgT, const bf16* __restrict__ wsuT,
                   const int* __restrict__ cnt, const int* __restrict__ offs,
                   const int* __restrict__ list, bf16* __restrict__ act) {
  __shared__ bf16 As[2][128][32];
  __shared__ bf16 Bg[2][64][32];
  __shared__ bf16 Bu[2][64][32];
  __shared__ int  tokS[128];

  int e  = blockIdx.y >> 5;     // 0..8 (8 == shared expert)
  int mt = blockIdx.y & 31;
  int n0 = blockIdx.x * 64;

  int rows, rbase;
  if (e < NEXP) { rows = cnt[e]; rbase = offs[e]; }
  else          { rows = TOKENS; rbase = 2 * TOKENS; }
  if (mt * 128 >= rows) return;

  const bf16* wgp = (e < NEXP) ? wgT + (size_t)e * DDIM * FDIM : wsgT;
  const bf16* wup = (e < NEXP) ? wuT + (size_t)e * DDIM * FDIM : wsuT;

  int tid = threadIdx.x;
  if (tid < 128) {
    int r = mt * 128 + tid;
    tokS[tid] = (e < NEXP) ? ((r < rows) ? list[rbase + r] : -1) : r;
  }
  __syncthreads();

  // staging decomposition: A 128x32 (512 16B chunks, 2/thread); B 64x32 (1/thread)
  int arow0 = tid >> 2, arow1 = arow0 + 64, aseg = tid & 3;
  int brow  = tid >> 2, bseg  = tid & 3;
  int tok0 = tokS[arow0], tok1 = tokS[arow1];
  const bf16* aP0 = xb + (size_t)(tok0 < 0 ? 0 : tok0) * DDIM + aseg * 8;
  const bf16* aP1 = xb + (size_t)(tok1 < 0 ? 0 : tok1) * DDIM + aseg * 8;
  const bf16* gP  = wgp + (size_t)(n0 + brow) * DDIM + bseg * 8;
  const bf16* uP  = wup + (size_t)(n0 + brow) * DDIM + bseg * 8;

  // wave fragment indices: 32(M) x 32(N) per wave
  int lane = tid & 31, wv = tid >> 5;
  int m0 = (wv & 3) * 32, wn = (wv >> 2) * 32;
  int fm = lane & 15, kb = (lane >> 4) * 8;

  v8f cg00 = {}, cg01 = {}, cg10 = {}, cg11 = {};
  v8f cu00 = {}, cu01 = {}, cu10 = {}, cu11 = {};

  const uint4 zz = {0, 0, 0, 0};

#if USE_ASYNC_LDS
  // padded A rows are never async-written: zero them once in both buffers
  if (tok0 < 0) { *(uint4*)&As[0][arow0][aseg * 8] = zz; *(uint4*)&As[1][arow0][aseg * 8] = zz; }
  if (tok1 < 0) { *(uint4*)&As[0][arow1][aseg * 8] = zz; *(uint4*)&As[1][arow1][aseg * 8] = zz; }
  if (tok0 >= 0) async_ld16(&As[0][arow0][aseg * 8], aP0);
  if (tok1 >= 0) async_ld16(&As[0][arow1][aseg * 8], aP1);
  async_ld16(&Bg[0][brow][bseg * 8], gP);
  async_ld16(&Bu[0][brow][bseg * 8], uP);
  WAIT_ASYNC();
  __syncthreads();
#else
  if (tok0 >= 0) *(v8bf*)&As[0][arow0][aseg * 8] = *(const v8bf*)(aP0);
  else           *(uint4*)&As[0][arow0][aseg * 8] = zz;
  if (tok1 >= 0) *(v8bf*)&As[0][arow1][aseg * 8] = *(const v8bf*)(aP1);
  else           *(uint4*)&As[0][arow1][aseg * 8] = zz;
  *(v8bf*)&Bg[0][brow][bseg * 8] = *(const v8bf*)(gP);
  *(v8bf*)&Bu[0][brow][bseg * 8] = *(const v8bf*)(uP);
  __syncthreads();
#endif

  int buf = 0;
  for (int k0 = 32; k0 < DDIM; k0 += 32) {
    int nb2 = buf ^ 1;
#if USE_ASYNC_LDS
    if (tok0 >= 0) async_ld16(&As[nb2][arow0][aseg * 8], aP0 + k0);
    if (tok1 >= 0) async_ld16(&As[nb2][arow1][aseg * 8], aP1 + k0);
    async_ld16(&Bg[nb2][brow][bseg * 8], gP + k0);
    async_ld16(&Bu[nb2][brow][bseg * 8], uP + k0);
#else
    v8bf ra0, ra1, rg, ru;
    ra0 = (tok0 >= 0) ? *(const v8bf*)(aP0 + k0) : *(const v8bf*)&zz;
    ra1 = (tok1 >= 0) ? *(const v8bf*)(aP1 + k0) : *(const v8bf*)&zz;
    rg = *(const v8bf*)(gP + k0);
    ru = *(const v8bf*)(uP + k0);
#endif
    {
      v16bf a0 = ldfrag(&As[buf][m0 + fm][kb]);
      v16bf a1 = ldfrag(&As[buf][m0 + 16 + fm][kb]);
      v16bf g0 = ldfrag(&Bg[buf][wn + fm][kb]);
      v16bf g1 = ldfrag(&Bg[buf][wn + 16 + fm][kb]);
      v16bf u0 = ldfrag(&Bu[buf][wn + fm][kb]);
      v16bf u1 = ldfrag(&Bu[buf][wn + 16 + fm][kb]);
      cg00 = WMMA_BF16(a0, g0, cg00); cg01 = WMMA_BF16(a0, g1, cg01);
      cg10 = WMMA_BF16(a1, g0, cg10); cg11 = WMMA_BF16(a1, g1, cg11);
      cu00 = WMMA_BF16(a0, u0, cu00); cu01 = WMMA_BF16(a0, u1, cu01);
      cu10 = WMMA_BF16(a1, u0, cu10); cu11 = WMMA_BF16(a1, u1, cu11);
    }
#if USE_ASYNC_LDS
    WAIT_ASYNC();
#else
    *(v8bf*)&As[nb2][arow0][aseg * 8] = ra0;
    *(v8bf*)&As[nb2][arow1][aseg * 8] = ra1;
    *(v8bf*)&Bg[nb2][brow][bseg * 8] = rg;
    *(v8bf*)&Bu[nb2][brow][bseg * 8] = ru;
#endif
    __syncthreads();
    buf = nb2;
  }
  {
    v16bf a0 = ldfrag(&As[buf][m0 + fm][kb]);
    v16bf a1 = ldfrag(&As[buf][m0 + 16 + fm][kb]);
    v16bf g0 = ldfrag(&Bg[buf][wn + fm][kb]);
    v16bf g1 = ldfrag(&Bg[buf][wn + 16 + fm][kb]);
    v16bf u0 = ldfrag(&Bu[buf][wn + fm][kb]);
    v16bf u1 = ldfrag(&Bu[buf][wn + 16 + fm][kb]);
    cg00 = WMMA_BF16(a0, g0, cg00); cg01 = WMMA_BF16(a0, g1, cg01);
    cg10 = WMMA_BF16(a1, g0, cg10); cg11 = WMMA_BF16(a1, g1, cg11);
    cu00 = WMMA_BF16(a0, u0, cu00); cu01 = WMMA_BF16(a0, u1, cu01);
    cu10 = WMMA_BF16(a1, u0, cu10); cu11 = WMMA_BF16(a1, u1, cu11);
  }

#pragma unroll
  for (int mi = 0; mi < 2; ++mi) {
    v8f g0v = mi ? cg10 : cg00, g1v = mi ? cg11 : cg01;
    v8f u0v = mi ? cu10 : cu00, u1v = mi ? cu11 : cu01;
#pragma unroll
    for (int r = 0; r < 8; ++r) {
      int m  = m0 + mi * 16 + r + ((lane >> 4) << 3);
      int gr = mt * 128 + m;
      if (gr < rows) {
        bf16* o = act + (size_t)(rbase + gr) * FDIM + n0 + wn + fm;
        o[0]  = (bf16)(silu_f(g0v[r]) * u0v[r]);
        o[16] = (bf16)(silu_f(g1v[r]) * u1v[r]);
      }
    }
  }
}

// ---------------------------------------------------------------- grouped GEMM 2
// out[token] += gate_weight * (act_row @ Wd); BM=128, BN=64, BK=32, 32x32/wave.
__global__ __launch_bounds__(256)
void k_gemm_down(const bf16* __restrict__ act,
                 const bf16* __restrict__ wdT, const bf16* __restrict__ wsdT,
                 const int* __restrict__ cnt, const int* __restrict__ offs,
                 const int* __restrict__ list, const float* __restrict__ wlist,
                 float* __restrict__ out) {
  __shared__ bf16  As[2][128][32];
  __shared__ bf16  Bd[2][64][32];
  __shared__ int   tokS[128];
  __shared__ float wS[128];

  int e  = blockIdx.y >> 5;
  int mt = blockIdx.y & 31;
  int n0 = blockIdx.x * 64;

  int rows, rbase;
  if (e < NEXP) { rows = cnt[e]; rbase = offs[e]; }
  else          { rows = TOKENS; rbase = 2 * TOKENS; }
  if (mt * 128 >= rows) return;

  const bf16* wdp = (e < NEXP) ? wdT + (size_t)e * FDIM * DDIM : wsdT;

  int tid = threadIdx.x;
  if (tid < 128) {
    int r = mt * 128 + tid;
    if (e < NEXP) {
      if (r < rows) { tokS[tid] = list[rbase + r]; wS[tid] = wlist[rbase + r]; }
      else          { tokS[tid] = -1;              wS[tid] = 0.0f; }
    } else          { tokS[tid] = r;               wS[tid] = 1.0f; }
  }
  __syncthreads();

  int arow0 = tid >> 2, arow1 = arow0 + 64, aseg = tid & 3;
  int brow  = tid >> 2, bseg  = tid & 3;
  const bf16* aP0 = act + (size_t)(rbase + mt * 128 + arow0) * FDIM + aseg * 8;
  const bf16* aP1 = act + (size_t)(rbase + mt * 128 + arow1) * FDIM + aseg * 8;
  const bf16* bP  = wdp + (size_t)(n0 + brow) * FDIM + bseg * 8;

  int lane = tid & 31, wv = tid >> 5;
  int m0 = (wv & 3) * 32, wn = (wv >> 2) * 32;
  int fm = lane & 15, kb = (lane >> 4) * 8;

  v8f c00 = {}, c01 = {}, c10 = {}, c11 = {};

#if USE_ASYNC_LDS
  async_ld16(&As[0][arow0][aseg * 8], aP0);
  async_ld16(&As[0][arow1][aseg * 8], aP1);
  async_ld16(&Bd[0][brow][bseg * 8], bP);
  WAIT_ASYNC();
  __syncthreads();
#else
  *(v8bf*)&As[0][arow0][aseg * 8] = *(const v8bf*)(aP0);
  *(v8bf*)&As[0][arow1][aseg * 8] = *(const v8bf*)(aP1);
  *(v8bf*)&Bd[0][brow][bseg * 8] = *(const v8bf*)(bP);
  __syncthreads();
#endif

  int buf = 0;
  for (int k0 = 32; k0 < FDIM; k0 += 32) {
    int nb2 = buf ^ 1;
#if USE_ASYNC_LDS
    async_ld16(&As[nb2][arow0][aseg * 8], aP0 + k0);
    async_ld16(&As[nb2][arow1][aseg * 8], aP1 + k0);
    async_ld16(&Bd[nb2][brow][bseg * 8], bP + k0);
#else
    v8bf ra0 = *(const v8bf*)(aP0 + k0);
    v8bf ra1 = *(const v8bf*)(aP1 + k0);
    v8bf rb  = *(const v8bf*)(bP + k0);
#endif
    {
      v16bf a0 = ldfrag(&As[buf][m0 + fm][kb]);
      v16bf a1 = ldfrag(&As[buf][m0 + 16 + fm][kb]);
      v16bf b0 = ldfrag(&Bd[buf][wn + fm][kb]);
      v16bf b1 = ldfrag(&Bd[buf][wn + 16 + fm][kb]);
      c00 = WMMA_BF16(a0, b0, c00); c01 = WMMA_BF16(a0, b1, c01);
      c10 = WMMA_BF16(a1, b0, c10); c11 = WMMA_BF16(a1, b1, c11);
    }
#if USE_ASYNC_LDS
    WAIT_ASYNC();
#else
    *(v8bf*)&As[nb2][arow0][aseg * 8] = ra0;
    *(v8bf*)&As[nb2][arow1][aseg * 8] = ra1;
    *(v8bf*)&Bd[nb2][brow][bseg * 8] = rb;
#endif
    __syncthreads();
    buf = nb2;
  }
  {
    v16bf a0 = ldfrag(&As[buf][m0 + fm][kb]);
    v16bf a1 = ldfrag(&As[buf][m0 + 16 + fm][kb]);
    v16bf b0 = ldfrag(&Bd[buf][wn + fm][kb]);
    v16bf b1 = ldfrag(&Bd[buf][wn + 16 + fm][kb]);
    c00 = WMMA_BF16(a0, b0, c00); c01 = WMMA_BF16(a0, b1, c01);
    c10 = WMMA_BF16(a1, b0, c10); c11 = WMMA_BF16(a1, b1, c11);
  }

#pragma unroll
  for (int mi = 0; mi < 2; ++mi) {
    v8f v0 = mi ? c10 : c00, v1 = mi ? c11 : c01;
#pragma unroll
    for (int r = 0; r < 8; ++r) {
      int m  = m0 + mi * 16 + r + ((lane >> 4) << 3);
      int tk = tokS[m];
      if (tk >= 0) {
        float w = wS[m];
        float* o = out + (size_t)tk * DDIM + n0 + wn + fm;
        __hip_atomic_fetch_add(o,      v0[r] * w, __ATOMIC_RELAXED, __HIP_MEMORY_SCOPE_AGENT);
        __hip_atomic_fetch_add(o + 16, v1[r] * w, __ATOMIC_RELAXED, __HIP_MEMORY_SCOPE_AGENT);
      }
    }
  }
}

// ---------------------------------------------------------------- host launcher

extern "C" void kernel_launch(void* const* d_in, const int* in_sizes, int n_in,
                              void* d_out, int out_size, void* d_ws, size_t ws_size,
                              hipStream_t stream) {
  (void)in_sizes; (void)n_in; (void)ws_size;
  const float* x       = (const float*)d_in[0];
  const float* gate_w  = (const float*)d_in[1];
  const float* w_gate  = (const float*)d_in[2];
  const float* w_up    = (const float*)d_in[3];
  const float* w_down  = (const float*)d_in[4];
  const float* ws_gate = (const float*)d_in[5];
  const float* ws_up   = (const float*)d_in[6];
  const float* ws_down = (const float*)d_in[7];
  float* out = (float*)d_out;

  char* p = (char*)d_ws;
  auto carve = [&](size_t bytes) -> void* {
    void* r = (void*)p;
    p += (bytes + 255) & ~(size_t)255;
    return r;
  };
  const size_t EDF = (size_t)NEXP * DDIM * FDIM;
  bf16*  xb    = (bf16*)carve((size_t)TOKENS * DDIM * 2);
  bf16*  wgT   = (bf16*)carve(EDF * 2);                     // [E][F][D]
  bf16*  wuT   = (bf16*)carve(EDF * 2);                     // [E][F][D]
  bf16*  wdT   = (bf16*)carve(EDF * 2);                     // [E][D][F]
  bf16*  wsgT  = (bf16*)carve((size_t)DDIM * FDIM * 2);     // [F][D]
  bf16*  wsuT  = (bf16*)carve((size_t)DDIM * FDIM * 2);     // [F][D]
  bf16*  wsdT  = (bf16*)carve((size_t)FDIM * DDIM * 2);     // [D][F]
  bf16*  act   = (bf16*)carve((size_t)3 * TOKENS * FDIM * 2);
  int*   tki   = (int*)carve((size_t)2 * TOKENS * 4);
  float* tkw   = (float*)carve((size_t)2 * TOKENS * 4);
  int*   list  = (int*)carve((size_t)2 * TOKENS * 4);
  float* wlist = (float*)carve((size_t)2 * TOKENS * 4);
  int*   cnt   = (int*)carve(NEXP * 4);
  int*   offs  = (int*)carve(NEXP * 4);
  int*   curs  = (int*)carve(NEXP * 4);

  k_zero_f32<<<(out_size + 255) / 256, 256, 0, stream>>>(out, out_size);
  (void)hipMemsetAsync(cnt, 0, NEXP * sizeof(int), stream);

  k_cvt_bf16<<<(unsigned)(((size_t)TOKENS * DDIM / 4 + 255) / 256), 256, 0, stream>>>(
      x, xb, TOKENS * DDIM);

  dim3 tb(32, 8);
  k_cvt_t<<<dim3(FDIM / 32, DDIM / 32, NEXP), tb, 0, stream>>>(w_gate,  wgT,  DDIM, FDIM);
  k_cvt_t<<<dim3(FDIM / 32, DDIM / 32, NEXP), tb, 0, stream>>>(w_up,    wuT,  DDIM, FDIM);
  k_cvt_t<<<dim3(DDIM / 32, FDIM / 32, NEXP), tb, 0, stream>>>(w_down,  wdT,  FDIM, DDIM);
  k_cvt_t<<<dim3(FDIM / 32, DDIM / 32, 1),    tb, 0, stream>>>(ws_gate, wsgT, DDIM, FDIM);
  k_cvt_t<<<dim3(FDIM / 32, DDIM / 32, 1),    tb, 0, stream>>>(ws_up,   wsuT, DDIM, FDIM);
  k_cvt_t<<<dim3(DDIM / 32, FDIM / 32, 1),    tb, 0, stream>>>(ws_down, wsdT, FDIM, DDIM);

  k_router<<<TOKENS / 256, 256, 0, stream>>>(x, gate_w, tki, tkw, cnt);
  k_scan<<<1, 64, 0, stream>>>(cnt, offs, curs);
  k_scatter<<<TOKENS / 256, 256, 0, stream>>>(tki, tkw, curs, list, wlist);

  // grouped GEMMs: grid.y = 9 groups (8 experts + shared) x 32 max M-tiles (BM=128)
  dim3 g1(FDIM / 64, 9 * 32);
  k_gemm_gateup<<<g1, 256, 0, stream>>>(xb, wgT, wuT, wsgT, wsuT, cnt, offs, list, act);

  dim3 g2(DDIM / 64, 9 * 32);
  k_gemm_down<<<g2, 256, 0, stream>>>(act, wdT, wsdT, cnt, offs, list, wlist, out);
}
